// Decoder_48644799594904
// MI455X (gfx1250) — matrix operand order repaired and checked by reading, not compiled, not run on hardware
//
#include <hip/hip_runtime.h>
#include <hip/hip_bf16.h>

// ---------------------------------------------------------------------------
// ConvS2S decoder forward for MI455X (gfx1250).
//  - all GEMMs through v_wmma_f32_16x16x32_bf16 (fp32 in/out, bf16 compute)
//  - A / transposed-B tiles staged with global_load_async_to_lds_b128
//    (ASYNCcnt) -- the gfx1250 async-copy path
//  - fragments built with ds_load_b128, stores/elementwise all b128
// ---------------------------------------------------------------------------

typedef __bf16 v16bf __attribute__((ext_vector_type(16)));
typedef float  v8f   __attribute__((ext_vector_type(8)));

#define EMB   256
#define HID   512
#define NLAY  3
#define KSZ   3
#define BB    8
#define TT    2048
#define SS    2048
#define LBIG  (-4294967295.0f)
#define RSQ2  0.70710678118654752440f

#define TILE_M 128
#define TILE_N 64
#define TILE_K 32
#define LDS_STRIDE 36   // floats; 144B rows: 16B-aligned, conflict-free mod 64 banks

// ---------------------------------------------------------------------------
// issue one async 16B global->LDS copy (per lane)
// ---------------------------------------------------------------------------
__device__ __forceinline__ void async_g2l_b128(void* lds_dst, const float* gsrc) {
  unsigned lds = (unsigned)(unsigned long long)lds_dst;          // low 32 bits = LDS offset
  asm volatile("global_load_async_to_lds_b128 %0, %1, off"
               :: "v"(lds), "v"((unsigned long long)gsrc)
               : "memory");
}
__device__ __forceinline__ void wait_async_all() {
  asm volatile("s_wait_asynccnt 0x0" ::: "memory");
}

__device__ __forceinline__ void cvt8(v16bf& f, int base, const float4 u, const float4 v) {
  f[base + 0] = (__bf16)u.x; f[base + 1] = (__bf16)u.y;
  f[base + 2] = (__bf16)u.z; f[base + 3] = (__bf16)u.w;
  f[base + 4] = (__bf16)v.x; f[base + 5] = (__bf16)v.y;
  f[base + 6] = (__bf16)v.z; f[base + 7] = (__bf16)v.w;
}

// ---------------------------------------------------------------------------
// Batched GEMM: C[b] = epi( A[b] @ B[b] + bias (+extra) )
//   A: [M,K] row-major (lda); B: [K,N] row-major (ldb), or [N,K] if BTRANS.
//   M%128 == N%64 == K%32 == 0 by construction -> no divergence around WMMA.
//   EPI: 0 = x+bias ; 1 = tanh(x+bias) ; 2 = (x+bias+extra)*scale
// ---------------------------------------------------------------------------
template <bool BTRANS, int EPI>
__global__ __launch_bounds__(256) void gemm_bf16_wmma(
    const float* __restrict__ A, const float* __restrict__ Bm,
    const float* __restrict__ bias, const float* __restrict__ extra,
    float* __restrict__ C,
    long sA, long sB, long sC, long sE,
    int lda, int ldb, int ldc, int lde,
    int Kdim, float scale)
{
  __shared__ __align__(16) float As [TILE_M][LDS_STRIDE];  // A tile, [m][k]
  __shared__ __align__(16) float Bst[TILE_N][LDS_STRIDE];  // B tile transposed, [n][k]

  const int bz = blockIdx.z;
  A  += (long)bz * sA;
  Bm += (long)bz * sB;
  C  += (long)bz * sC;
  if (EPI == 2) extra += (long)bz * sE;

  const int m0   = blockIdx.x * TILE_M;
  const int n0   = blockIdx.y * TILE_N;
  const int tid  = threadIdx.x;
  const int lane = tid & 31;
  const int wave = tid >> 5;
  const int wm   = wave & 3;   // 4 waves along M (32 rows each)
  const int wn   = wave >> 2;  // 2 waves along N (32 cols each)
  const int lm   = lane & 15;
  const int lh   = lane >> 4;

  v8f acc[2][2];
  acc[0][0] = (v8f)0.0f; acc[0][1] = (v8f)0.0f;
  acc[1][0] = (v8f)0.0f; acc[1][1] = (v8f)0.0f;

  for (int k0 = 0; k0 < Kdim; k0 += TILE_K) {
    // ---- A tile 128x32: 1024 x float4, 4 async b128 copies per thread ----
#pragma unroll
    for (int j = 0; j < 4; ++j) {
      int q   = tid + 256 * j;
      int row = q >> 3;
      int c4  = (q & 7) << 2;
      async_g2l_b128(&As[row][c4], A + (long)(m0 + row) * lda + (k0 + c4));
    }
    if (BTRANS) {
      // B stored [N,K]: rows map straight into Bst -> async b128
#pragma unroll
      for (int j = 0; j < 2; ++j) {
        int q  = tid + 256 * j;
        int n  = q >> 3;
        int c4 = (q & 7) << 2;
        async_g2l_b128(&Bst[n][c4], Bm + (long)(n0 + n) * ldb + (k0 + c4));
      }
    } else {
      // B stored [K,N]: b128 gather along N, transpose-scatter into Bst
      float4 rb[2]; int bk[2], bn[2];
#pragma unroll
      for (int j = 0; j < 2; ++j) {
        int q  = tid + 256 * j;
        bk[j]  = q >> 4;
        bn[j]  = (q & 15) << 2;
        rb[j]  = *(const float4*)(Bm + (long)(k0 + bk[j]) * ldb + (n0 + bn[j]));
      }
#pragma unroll
      for (int j = 0; j < 2; ++j) {
        Bst[bn[j] + 0][bk[j]] = rb[j].x;
        Bst[bn[j] + 1][bk[j]] = rb[j].y;
        Bst[bn[j] + 2][bk[j]] = rb[j].z;
        Bst[bn[j] + 3][bk[j]] = rb[j].w;
      }
    }
    // prefetch next A tile (global_prefetch_b8)
    __builtin_prefetch(A + (long)(m0 + (tid >> 5)) * lda + (k0 + TILE_K) + (tid & 31), 0, 0);

    wait_async_all();
    __syncthreads();

    // ---- fragments: 4x ds_load_b128 each, cvt to bf16 ----
    v16bf afrag[2], bfrag[2];
#pragma unroll
    for (int mi = 0; mi < 2; ++mi) {
      int row = wm * 32 + mi * 16 + lm;
      const float* p = &As[row][0];
      float4 u0 = *(const float4*)(p + 8 * lh);
      float4 u1 = *(const float4*)(p + 8 * lh + 4);
      float4 u2 = *(const float4*)(p + 16 + 8 * lh);
      float4 u3 = *(const float4*)(p + 16 + 8 * lh + 4);
      cvt8(afrag[mi], 0, u0, u1);
      cvt8(afrag[mi], 8, u2, u3);
    }
#pragma unroll
    for (int ni = 0; ni < 2; ++ni) {
      int col = wn * 32 + ni * 16 + lm;
      const float* p = &Bst[col][8 * lh];
      float4 u0 = *(const float4*)(p + 0);
      float4 u1 = *(const float4*)(p + 4);
      float4 u2 = *(const float4*)(p + 16);
      float4 u3 = *(const float4*)(p + 20);
      cvt8(bfrag[ni], 0, u0, u1);
      cvt8(bfrag[ni], 8, u2, u3);
    }
#pragma unroll
    for (int mi = 0; mi < 2; ++mi)
#pragma unroll
      for (int ni = 0; ni < 2; ++ni)
        acc[mi][ni] = __builtin_amdgcn_wmma_f32_16x16x32_bf16(
            false, afrag[mi], false, bfrag[ni],
            (short)0, acc[mi][ni], false, false);
    __syncthreads();
  }

  // ---- epilogue + store (C/D layout: VGPR r -> M = 8*lh + r) ----
#pragma unroll
  for (int mi = 0; mi < 2; ++mi) {
#pragma unroll
    for (int ni = 0; ni < 2; ++ni) {
      int col = n0 + wn * 32 + ni * 16 + lm;
      float bv = bias ? bias[col] : 0.0f;
#pragma unroll
      for (int r = 0; r < 8; ++r) {
        int row = m0 + wm * 32 + mi * 16 + lh * 8 + r;
        float v = acc[mi][ni][r] + bv;
        if (EPI == 1) v = tanhf(v);
        if (EPI == 2) v = (v + extra[(long)row * lde + col]) * scale;
        C[(long)row * ldc + col] = v;
      }
    }
  }
}

// ---------------------------------------------------------------------------
// Small helper kernels (all float4 / b128)
// ---------------------------------------------------------------------------
__global__ void k_padproj(const float* __restrict__ pad_vec,
                          const float* __restrict__ W_e2h,
                          const float* __restrict__ b_e2h,
                          float* __restrict__ padproj)
{
  int h = blockIdx.x * 256 + threadIdx.x;
  if (h >= HID) return;
  float s = b_e2h[h];
  for (int e = 0; e < EMB; ++e) s += pad_vec[e] * W_e2h[e * HID + h];
  padproj[h] = tanhf(s);
}

__global__ void k_padfill(const float* __restrict__ padproj, float* __restrict__ xpad)
{
  int idx = blockIdx.x * 256 + threadIdx.x;            // BB*2*HID/4 = 2048
  if (idx >= BB * 2 * (HID / 4)) return;
  int h4 = idx & (HID / 4 - 1);
  int r  = (idx >> 7) & 1;
  int b  = idx >> 8;
  ((float4*)xpad)[(long)b * ((TT + 2) * HID / 4) + r * (HID / 4) + h4] =
      ((const float4*)padproj)[h4];
}

__global__ void k_embed(const float* __restrict__ trg, const float* __restrict__ pos_emb,
                        float* __restrict__ embedded)
{
  long i = (long)blockIdx.x * 256 + threadIdx.x;       // BB*TT*EMB/4
  if (i >= (long)BB * TT * EMB / 4) return;
  float4 a = ((const float4*)trg)[i];
  float4 p = ((const float4*)pos_emb)[i % ((long)TT * EMB / 4)];
  float4 o; o.x = a.x + p.x; o.y = a.y + p.y; o.z = a.z + p.z; o.w = a.w + p.w;
  ((float4*)embedded)[i] = o;
}

// conv_w[i] (O=1024, H=512, K=3) -> Wt[(k*512+h), o] row-major [1536,1024]
__global__ void k_repack(const float* __restrict__ cw, float* __restrict__ Wt)
{
  long i = (long)blockIdx.x * 256 + threadIdx.x;       // 1536*1024/4
  if (i >= (long)KSZ * HID * 2 * HID / 4) return;
  int o4 = (int)(i & 255) << 2;
  int r  = (int)(i >> 8);
  int k  = r / HID;
  int h  = r - k * HID;
  float4 v;
  v.x = cw[(long)(o4 + 0) * (HID * KSZ) + h * KSZ + k];
  v.y = cw[(long)(o4 + 1) * (HID * KSZ) + h * KSZ + k];
  v.z = cw[(long)(o4 + 2) * (HID * KSZ) + h * KSZ + k];
  v.w = cw[(long)(o4 + 3) * (HID * KSZ) + h * KSZ + k];
  ((float4*)Wt)[i] = v;
}

__device__ __forceinline__ float sigf(float x) { return 1.0f / (1.0f + expf(-x)); }

__global__ void k_glu(const float* __restrict__ cbuf, float* __restrict__ conved)
{
  long i = (long)blockIdx.x * 256 + threadIdx.x;       // BB*TT*HID/4
  if (i >= (long)BB * TT * HID / 4) return;
  long m  = i >> 7;                                    // row (b*T+t), HID/4=128
  int  h4 = (int)(i & 127);
  float4 a = ((const float4*)cbuf)[m * (2 * HID / 4) + h4];
  float4 g = ((const float4*)cbuf)[m * (2 * HID / 4) + (HID / 4) + h4];
  float4 o;
  o.x = a.x * sigf(g.x); o.y = a.y * sigf(g.y);
  o.z = a.z * sigf(g.z); o.w = a.w * sigf(g.w);
  ((float4*)conved)[i] = o;
}

// conved=(conved+aenc)*s ; conv_input(next) = (conved+conv_input)*s   (xpad)
__global__ void k_update(const float* __restrict__ conved, const float* __restrict__ aenc,
                         float* __restrict__ xpad)
{
  long i = (long)blockIdx.x * 256 + threadIdx.x;       // BB*TT*HID/4
  if (i >= (long)BB * TT * HID / 4) return;
  long m  = i >> 7;
  int  h4 = (int)(i & 127);
  int  t  = (int)(m & (TT - 1));
  int  b  = (int)(m >> 11);
  long xi = (long)b * ((TT + 2) * HID / 4) + (long)(t + 2) * (HID / 4) + h4;
  float4 cv = ((const float4*)conved)[i];
  float4 ae = ((const float4*)aenc)[i];
  float4 xp = ((float4*)xpad)[xi];
  float4 o;
  o.x = ((cv.x + ae.x) * RSQ2 + xp.x) * RSQ2;
  o.y = ((cv.y + ae.y) * RSQ2 + xp.y) * RSQ2;
  o.z = ((cv.z + ae.z) * RSQ2 + xp.z) * RSQ2;
  o.w = ((cv.w + ae.w) * RSQ2 + xp.w) * RSQ2;
  ((float4*)xpad)[xi] = o;
}

// masked softmax over S, in place; one block per (b,t) row, float4 passes
__global__ __launch_bounds__(256) void k_softmax(float* __restrict__ attn,
                                                 const float* __restrict__ mask)
{
  __shared__ float red[256];
  int row = blockIdx.x;                  // b*TT + t
  int b   = row >> 11;
  float4* x = (float4*)(attn + (long)row * SS);
  const float4* mk = (const float4*)(mask + (long)b * SS);

  float mx = -3.4e38f;
  for (int s = threadIdx.x; s < SS / 4; s += 256) {
    float4 v = x[s];
    float4 m4 = mk[s];
    v.x += (1.0f - m4.x) * LBIG; v.y += (1.0f - m4.y) * LBIG;
    v.z += (1.0f - m4.z) * LBIG; v.w += (1.0f - m4.w) * LBIG;
    x[s] = v;
    mx = fmaxf(mx, fmaxf(fmaxf(v.x, v.y), fmaxf(v.z, v.w)));
  }
  red[threadIdx.x] = mx; __syncthreads();
  for (int off = 128; off > 0; off >>= 1) {
    if (threadIdx.x < off) red[threadIdx.x] = fmaxf(red[threadIdx.x], red[threadIdx.x + off]);
    __syncthreads();
  }
  mx = red[0]; __syncthreads();

  float sum = 0.0f;
  for (int s = threadIdx.x; s < SS / 4; s += 256) {
    float4 v = x[s];
    v.x = expf(v.x - mx); v.y = expf(v.y - mx);
    v.z = expf(v.z - mx); v.w = expf(v.w - mx);
    x[s] = v;
    sum += v.x + v.y + v.z + v.w;
  }
  red[threadIdx.x] = sum; __syncthreads();
  for (int off = 128; off > 0; off >>= 1) {
    if (threadIdx.x < off) red[threadIdx.x] += red[threadIdx.x + off];
    __syncthreads();
  }
  float inv = 1.0f / red[0];
  for (int s = threadIdx.x; s < SS / 4; s += 256) {
    float4 v = x[s];
    v.x *= inv; v.y *= inv; v.z *= inv; v.w *= inv;
    x[s] = v;
  }
}

// ---------------------------------------------------------------------------
// Host-side orchestration
// ---------------------------------------------------------------------------
template <bool BT, int EPI>
static inline void launch_gemm(hipStream_t st, int gm, int gn, int gb,
                               const float* A, const float* Bm,
                               const float* bias, const float* extra, float* C,
                               long sA, long sB, long sC, long sE,
                               int lda, int ldb, int ldc, int lde,
                               int K, float scale)
{
  dim3 grid(gm, gn, gb);
  gemm_bf16_wmma<BT, EPI><<<grid, 256, 0, st>>>(A, Bm, bias, extra, C,
                                                sA, sB, sC, sE,
                                                lda, ldb, ldc, lde, K, scale);
}

extern "C" void kernel_launch(void* const* d_in, const int* in_sizes, int n_in,
                              void* d_out, int out_size, void* d_ws, size_t ws_size,
                              hipStream_t stream) {
  (void)in_sizes; (void)n_in; (void)out_size; (void)ws_size;
  const float* trg        = (const float*)d_in[0];
  const float* src        = (const float*)d_in[1];
  const float* enc_conved = (const float*)d_in[2];
  const float* enc_embed  = (const float*)d_in[3];
  const float* mask       = (const float*)d_in[4];
  const float* pos_emb    = (const float*)d_in[5];
  const float* pad_vec    = (const float*)d_in[6];
  const float* W_e2h      = (const float*)d_in[7];
  const float* b_e2h      = (const float*)d_in[8];
  const float* W_ah2e     = (const float*)d_in[9];
  const float* b_ah2e     = (const float*)d_in[10];
  const float* W_ae2h     = (const float*)d_in[11];
  const float* b_ae2h     = (const float*)d_in[12];
  const float* W_out      = (const float*)d_in[13];
  const float* b_out      = (const float*)d_in[14];
  const float* conv_w     = (const float*)d_in[15];
  const float* conv_b     = (const float*)d_in[16];

  const long N_BTE  = (long)BB * TT * EMB;       //  4,194,304
  const long N_BTH  = (long)BB * TT * HID;       //  8,388,608
  const long XPAD_B = (long)(TT + 2) * HID;      //  1,049,600 per batch
  float* ws       = (float*)d_ws;
  float* padproj  = ws;                                   // 512
  float* xpad     = ws + 512;                             // BB*XPAD_B
  float* Wt       = xpad + BB * XPAD_B;                   // 1536*1024
  float* cbuf     = Wt + (long)KSZ * HID * 2 * HID;       // BB*TT*2H
  float* embedded = cbuf;                                 // aliases cbuf (disjoint lifetime)
  float* conved   = cbuf + (long)BB * TT * 2 * HID;       // BB*TT*H
  float* cemb     = conved + N_BTH;                       // BB*TT*E
  float* actx     = cemb + N_BTE;                         // BB*TT*E
  float* aenc     = actx + N_BTE;                         // BB*TT*H
  float* asrc     = aenc + N_BTH;                         // BB*TT*E

  float* outp = (float*)d_out;                            // [B,T,E]
  float* attn = outp + N_BTE;                             // [B,T,S]

  // 1) pad projection + fill xpad[:, 0:2, :]
  k_padproj<<<2, 256, 0, stream>>>(pad_vec, W_e2h, b_e2h, padproj);
  k_padfill<<<(BB * 2 * (HID / 4) + 255) / 256, 256, 0, stream>>>(padproj, xpad);

  // 2) embedded = trg + pos_emb
  k_embed<<<(int)(N_BTE / 4 / 256), 256, 0, stream>>>(trg, pos_emb, embedded);

  // 3) conv_input = tanh(embedded @ W_e2h + b) -> xpad[:, 2:, :]
  launch_gemm<false, 1>(stream, TT / TILE_M, HID / TILE_N, BB,
                        embedded, W_e2h, b_e2h, nullptr, xpad + 2 * HID,
                        (long)TT * EMB, 0, XPAD_B, 0,
                        EMB, HID, HID, 0, EMB, 1.0f);

  for (int i = 0; i < NLAY; ++i) {
    // a) repack conv weights -> [1536,1024]
    k_repack<<<(KSZ * HID * 2 * HID / 4) / 256, 256, 0, stream>>>(
        conv_w + (long)i * 2 * HID * HID * KSZ, Wt);

    // b) conv as GEMM: M=T, N=2H, K=3H (A rows overlap inside xpad, lda=H)
    launch_gemm<false, 0>(stream, TT / TILE_M, (2 * HID) / TILE_N, BB,
                          xpad, Wt, conv_b + (long)i * 2 * HID, nullptr, cbuf,
                          XPAD_B, 0, (long)TT * 2 * HID, 0,
                          HID, 2 * HID, 2 * HID, 0, KSZ * HID, 1.0f);

    // c) GLU
    k_glu<<<(int)(N_BTH / 4 / 256), 256, 0, stream>>>(cbuf, conved);

    // d) conved_emb = tanh(conved @ W_ah2e + b)   (flatten batch)
    launch_gemm<false, 1>(stream, (BB * TT) / TILE_M, EMB / TILE_N, 1,
                          conved, W_ah2e, b_ah2e, nullptr, cemb,
                          0, 0, 0, 0,
                          HID, EMB, EMB, 0, HID, 1.0f);

    // e) energy = cemb @ enc_conved^T (batched, B-trans) -> attn slot in d_out
    launch_gemm<true, 0>(stream, TT / TILE_M, SS / TILE_N, BB,
                         cemb, enc_conved, nullptr, nullptr, attn,
                         (long)TT * EMB, (long)SS * EMB, (long)TT * SS, 0,
                         EMB, EMB, SS, 0, EMB, 1.0f);

    // f) masked softmax over S (in place)
    k_softmax<<<BB * TT, 256, 0, stream>>>(attn, mask);

    // g) ctx = attn @ enc_embed (batched)
    launch_gemm<false, 0>(stream, TT / TILE_M, EMB / TILE_N, BB,
                          attn, enc_embed, nullptr, nullptr, actx,
                          (long)TT * SS, (long)SS * EMB, (long)TT * EMB, 0,
                          SS, EMB, EMB, 0, SS, 1.0f);

    // h) att_enc = tanh(ctx @ W_ae2h + b)  (flatten batch)
    launch_gemm<false, 1>(stream, (BB * TT) / TILE_M, HID / TILE_N, 1,
                          actx, W_ae2h, b_ae2h, nullptr, aenc,
                          0, 0, 0, 0,
                          EMB, HID, HID, 0, EMB, 1.0f);

    // i) residual updates into xpad (next conv_input / final conved)
    k_update<<<(int)(N_BTH / 4 / 256), 256, 0, stream>>>(conved, aenc, xpad);
  }

  // 4) attended_src = attn @ src (batched)
  launch_gemm<false, 0>(stream, TT / TILE_M, EMB / TILE_N, BB,
                        attn, src, nullptr, nullptr, asrc,
                        (long)TT * SS, (long)SS * EMB, (long)TT * EMB, 0,
                        SS, EMB, EMB, 0, SS, 1.0f);

  // 5) output = (conv_input @ W_out + b_out + attended_src) * SCALE
  launch_gemm<false, 2>(stream, TT / TILE_M, EMB / TILE_N, BB,
                        xpad + 2 * HID, W_out, b_out, asrc, outp,
                        XPAD_B, 0, (long)TT * EMB, (long)TT * EMB,
                        HID, EMB, EMB, EMB, HID, RSQ2);
}
